// self_attention_9225589752302
// MI455X (gfx1250) — matrix operand order repaired and checked by reading, not compiled, not run on hardware
//
#include <hip/hip_runtime.h>

#define NTOK  1024
#define CDIM  64
#define BQ    128
#define TJ    64
#define NSTEP (NTOK / TJ)
#define LOG2E 1.44269504088896340736f

typedef __attribute__((ext_vector_type(16))) __bf16 v16bf;
typedef __attribute__((ext_vector_type(4)))  __bf16 v4bf;
typedef __attribute__((ext_vector_type(8)))  float  v8f;
typedef unsigned int v4u __attribute__((ext_vector_type(4)));

#define LSTRIDE 72   // bf16 elems per LDS row (64 + 8 pad) = 144 B, multiple of 16 B

// LDS map (bytes):
//   region A @0     : prologue Q staging [128][72] bf16 = 18432 B
//                     reused in the loop as buffer 1: J1 @0, V1 @9216
//   buffer 0 @18432 : J0 @18432, V0 @27648 (9216 B each)
#define J_OFF(b)   ((b) ? 0u    : 18432u)
#define V_OFF(b)   ((b) ? 9216u : 27648u)
#define SMEM_BYTES 36864

// A-operand 16x32 bf16 (row M = lane&15): K = hi*16..+15 contiguous in the LDS row
static __device__ inline v16bf lds_a16x32(const __bf16* row, int hi) {
    union U { v4u u[2]; v16bf v; } t;
    t.u[0] = *(const v4u*)(row + hi * 16);
    t.u[1] = *(const v4u*)(row + hi * 16 + 8);
    return t.v;
}
// B-operand 32x16 bf16 (col N = lane&15): K = hi*16..+15 contiguous in the LDS row
static __device__ inline v16bf lds_b32x16(const __bf16* col, int hi) {
    union U { v4u u[2]; v16bf v; } t;
    t.u[0] = *(const v4u*)(col + hi * 16);
    t.u[1] = *(const v4u*)(col + hi * 16 + 8);
    return t.v;
}

__global__ __launch_bounds__(256) void fattn_kernel(const float* __restrict__ Kset,
                                                    const float* __restrict__ Jset,
                                                    const float* __restrict__ Vset,
                                                    float* __restrict__ Out) {
    __shared__ __align__(16) unsigned char smem[SMEM_BYTES];

    const int t    = threadIdx.x;
    const int lane = t & 31;
    const int wave = t >> 5;
    const int lh   = lane & 15;
    const int hi   = lane >> 4;
    const int head = blockIdx.y;
    const int q0g  = blockIdx.x * BQ;
    const size_t base = (size_t)head * CDIM * NTOK;

    // row-permutation base for S^T tiles: row M -> j offset (M&7) + 16*(M>>3)
    const int rbase = (lh & 7) + ((lh >> 3) << 4);

    // ---- prologue: stage Q (from K_set) as bf16 [q][c], and tile 0 into buffer 0 ----
    __bf16* QA = (__bf16*)smem;
#pragma unroll
    for (int i = 0; i < 8; ++i) {
        int idx = i * 1024 + t * 4;
        int q = idx & (BQ - 1);            // 4 consecutive q -> float4 global load
        int c = idx >> 7;
        float4 v = *(const float4*)(Kset + base + (size_t)c * NTOK + q0g + q);
        QA[(q + 0) * LSTRIDE + c] = (__bf16)v.x;
        QA[(q + 1) * LSTRIDE + c] = (__bf16)v.y;
        QA[(q + 2) * LSTRIDE + c] = (__bf16)v.z;
        QA[(q + 3) * LSTRIDE + c] = (__bf16)v.w;
    }
    {
        __bf16* Jl = (__bf16*)(smem + J_OFF(0));
        __bf16* Vl = (__bf16*)(smem + V_OFF(0));
#pragma unroll
        for (int i = 0; i < 4; ++i) {
            int idx = i * 1024 + t * 4;
            int jj = idx & (TJ - 1);
            int c  = idx >> 6;
            size_t g = base + (size_t)c * NTOK + jj;
            float4 jv = *(const float4*)(Jset + g);
            float4 vv = *(const float4*)(Vset + g);
            Jl[(jj + 0) * LSTRIDE + c] = (__bf16)jv.x;
            Jl[(jj + 1) * LSTRIDE + c] = (__bf16)jv.y;
            Jl[(jj + 2) * LSTRIDE + c] = (__bf16)jv.z;
            Jl[(jj + 3) * LSTRIDE + c] = (__bf16)jv.w;
            v4bf pv = { (__bf16)vv.x, (__bf16)vv.y, (__bf16)vv.z, (__bf16)vv.w };
            *(v4bf*)(Vl + c * LSTRIDE + jj) = pv;
        }
    }
    __syncthreads();

    // fixed B operands: Q^T [32c x 16q] per c-half, straight from Ql rows
    const __bf16* qrow = QA + (wave * 16 + lh) * LSTRIDE;
    v16bf bQ0 = lds_b32x16(qrow, hi);        // c = 0..31
    v16bf bQ1 = lds_b32x16(qrow + 32, hi);   // c = 32..63
    asm volatile("s_wait_dscnt 0x0" ::: "memory");
    __syncthreads();                         // region A may now become buffer 1

    v8f O[4] = {v8f{}, v8f{}, v8f{}, v8f{}};
    float m2 = -3.0e38f;
    float l  = 0.0f;

    for (int s = 0; s < NSTEP; ++s) {
        const int buf = s & 1;
        const __bf16* Jl = (const __bf16*)(smem + J_OFF(buf));
        const __bf16* Vl = (const __bf16*)(smem + V_OFF(buf));

        // ---- issue next tile's global loads before compute (double buffering) ----
        float4 gJ[4], gV[4];
        if (s + 1 < NSTEP) {
#pragma unroll
            for (int i = 0; i < 4; ++i) {
                int idx = i * 1024 + t * 4;
                int jj = idx & (TJ - 1);
                int c  = idx >> 6;
                size_t g = base + (size_t)c * NTOK + (s + 1) * TJ + jj;
                gJ[i] = *(const float4*)(Jset + g);
                gV[i] = *(const float4*)(Vset + g);
            }
            if (s + 2 < NSTEP) {
                size_t nxt = base + (size_t)(t >> 2) * NTOK + (s + 2) * TJ + (t & 3) * 16;
                __builtin_prefetch(Jset + nxt, 0, 3);
                __builtin_prefetch(Vset + nxt, 0, 3);
            }
        }

        // ---- S^T = J · Q^T with permuted rows:
        //      tile(jk,u): lane(lh,hi) holds j = 32*jk + 8*u + r + 16*hi  (r = VGPR idx), q = lh
        v8f S[2][2];
        float mt[4];                         // 4 independent max chains (one per tile)
#pragma unroll
        for (int jk = 0; jk < 2; ++jk)
#pragma unroll
            for (int u = 0; u < 2; ++u) {
                const __bf16* jrow = Jl + (jk * 32 + u * 8 + rbase) * LSTRIDE;
                v16bf aJ0 = lds_a16x32(jrow, hi);
                v16bf aJ1 = lds_a16x32(jrow + 32, hi);
                v8f acc = {};
                acc = __builtin_amdgcn_wmma_f32_16x16x32_bf16(false, aJ0, false, bQ0,
                                                              (short)0, acc, false, false);
                acc = __builtin_amdgcn_wmma_f32_16x16x32_bf16(false, aJ1, false, bQ1,
                                                              (short)0, acc, false, false);
                S[jk][u] = acc;
                // per-tile max as its own short chain (independent of other tiles)
                float a0 = fmaxf(fmaxf(acc[0], acc[1]), fmaxf(acc[2], acc[3]));
                float a1 = fmaxf(fmaxf(acc[4], acc[5]), fmaxf(acc[6], acc[7]));
                mt[jk * 2 + u] = fmaxf(a0, a1);
            }

        // ---- online softmax over j: combine 4 chains + one half-swap ----
        float mp = fmaxf(fmaxf(mt[0], mt[1]), fmaxf(mt[2], mt[3]));
        mp = fmaxf(mp, __shfl_xor(mp, 16));
        float m2n   = fmaxf(m2, mp * LOG2E);
        float scale = __builtin_amdgcn_exp2f(m2 - m2n);
        m2 = m2n;

#pragma unroll
        for (int cb = 0; cb < 4; ++cb) O[cb] = O[cb] * scale;

        // ---- exponentiate; P^T B-operand assembles directly (no shuffles/selects) ----
        float lp0 = 0.0f, lp1 = 0.0f;        // 2 independent sum chains
#pragma unroll
        for (int jk = 0; jk < 2; ++jk) {
            v16bf bP;
#pragma unroll
            for (int u = 0; u < 2; ++u)
#pragma unroll
                for (int r = 0; r < 8; ++r) {
                    float e = __builtin_amdgcn_exp2f(S[jk][u][r] * LOG2E - m2n);
                    if (r & 1) lp1 += e; else lp0 += e;
                    bP[u * 8 + r] = (__bf16)e;   // K = 16*hi + 8*u + r  ==  this lane's j
                }
#pragma unroll
            for (int cb = 0; cb < 4; ++cb) {
                const __bf16* vrow = Vl + (cb * 16 + lh) * LSTRIDE + jk * 32;
                v16bf aV = lds_a16x32(vrow, hi);
                O[cb] = __builtin_amdgcn_wmma_f32_16x16x32_bf16(false, aV, false, bP,
                                                                (short)0, O[cb], false, false);
            }
        }
        float lpart = lp0 + lp1;
        l = l * scale + lpart + __shfl_xor(lpart, 16);

        // ---- convert + store next tile into the other buffer ----
        if (s + 1 < NSTEP) {
            __bf16* Jn = (__bf16*)(smem + J_OFF(buf ^ 1));
            __bf16* Vn = (__bf16*)(smem + V_OFF(buf ^ 1));
#pragma unroll
            for (int i = 0; i < 4; ++i) {
                int idx = i * 1024 + t * 4;
                int jj = idx & (TJ - 1);
                int c  = idx >> 6;
                Jn[(jj + 0) * LSTRIDE + c] = (__bf16)gJ[i].x;
                Jn[(jj + 1) * LSTRIDE + c] = (__bf16)gJ[i].y;
                Jn[(jj + 2) * LSTRIDE + c] = (__bf16)gJ[i].z;
                Jn[(jj + 3) * LSTRIDE + c] = (__bf16)gJ[i].w;
                v4bf pv = { (__bf16)gV[i].x, (__bf16)gV[i].y, (__bf16)gV[i].z, (__bf16)gV[i].w };
                *(v4bf*)(Vn + c * LSTRIDE + jj) = pv;
            }
        }
        __syncthreads();
    }

    // ---- epilogue: O^T already has q across lanes -> direct coalesced stores ----
    float rl = 1.0f / l;
#pragma unroll
    for (int cb = 0; cb < 4; ++cb)
#pragma unroll
        for (int r = 0; r < 8; ++r) {
            int c = cb * 16 + r + hi * 8;
            Out[base + (size_t)c * NTOK + q0g + wave * 16 + lh] = O[cb][r] * rl;
        }
}

extern "C" void kernel_launch(void* const* d_in, const int* in_sizes, int n_in,
                              void* d_out, int out_size, void* d_ws, size_t ws_size,
                              hipStream_t stream) {
    const float* Kset = (const float*)d_in[0];   // queries
    const float* Jset = (const float*)d_in[1];   // keys
    const float* Vset = (const float*)d_in[2];   // values
    float* out = (float*)d_out;
    dim3 grid(NTOK / BQ, 64);                    // 8 query tiles x 64 heads
    fattn_kernel<<<grid, 256, 0, stream>>>(Kset, Jset, Vset, out);
}
